// MVM_Model_76510547411494
// MI455X (gfx1250) — 8-bit matrix operand order repaired and checked by reading, compiled, not run on hardware
//
#include <hip/hip_runtime.h>

typedef __attribute__((ext_vector_type(8))) int v8i;

#define BATCH 1024

__device__ __forceinline__ short quant16_13(float x) {
  return (short)(int)fminf(fmaxf(rintf(x * 8192.0f), -32768.0f), 32767.0f);
}

__global__ void normalize_kernel(const float* __restrict__ x, float* __restrict__ y,
                                 short* __restrict__ qy, int total) {
  int i = blockIdx.x * blockDim.x + threadIdx.x;
  if (i >= total) return;
  const float mean[3] = {0.485f, 0.456f, 0.406f};
  const float stdv[3] = {0.229f, 0.224f, 0.225f};
  int c = (i >> 10) % 3;
  float v = (x[i] - mean[c]) / stdv[c];
  y[i] = v;
  qy[i] = quant16_13(v);
}

__global__ void bn_prep_kernel(const float* __restrict__ g, const float* __restrict__ b,
                               const float* __restrict__ m, const float* __restrict__ v,
                               float* __restrict__ inv, float* __restrict__ sh, int C) {
  int i = blockIdx.x * blockDim.x + threadIdx.x;
  if (i >= C) return;
  float iv = g[i] / sqrtf(v[i] + 1e-5f);
  inv[i] = iv;
  sh[i] = b[i] - m[i] * iv;
}

__global__ void pack_weights_kernel(const float* __restrict__ w, int Cout, int K, int K64,
                                    int* __restrict__ a_hi, int* __restrict__ a_lo) {
  int idx = blockIdx.x * blockDim.x + threadIdx.x;
  int total = (Cout >> 4) * K64 * 256;
  if (idx >= total) return;
  int r = idx & 7;
  int lane = (idx >> 3) & 31;
  int rest = idx >> 8;
  int kb = rest % K64;
  int mt = rest / K64;
  int m = lane & 15, hsel = lane >> 4;
  int row = mt * 16 + m;
  int kbase = kb * 64 + ((r & 1) * 4) + ((r >> 1) * 16) + hsel * 8;
  int hi = 0, lo = 0;
#pragma unroll
  for (int j = 0; j < 4; ++j) {
    int k = kbase + j;
    int q = 0;
    if (k < K) q = (int)quant16_13(w[row * K + k]);
    hi |= ((q >> 8) & 0xFF) << (8 * j);
    lo |= (q & 0xFF) << (8 * j);
  }
  a_hi[idx] = hi;
  a_lo[idx] = lo;
}

template <int F>
__global__ __launch_bounds__(128) void conv_bn_wmma_kernel(
    const short* __restrict__ qin, const int* __restrict__ a_hi, const int* __restrict__ a_lo,
    const float* __restrict__ bn_inv, const float* __restrict__ bn_sh,
    const float* __restrict__ resid, float* __restrict__ outp, short* __restrict__ qout,
    int Cin, int Cout, int H, int W, int woShift, int howoShift,
    int stride, int pad, int K, int K64, int do_relu) {
  constexpr int FF = F * F;
  __shared__ int sHi[512];
  __shared__ int sLo[512];
  const int tid = threadIdx.x;
  const int lane = tid & 31;
  const int mtile = tid >> 5;
  const int ntile = blockIdx.x;
  const int nthreads = blockDim.x;
  const int Wmask = (1 << woShift) - 1;
  const int HoWomask = (1 << howoShift) - 1;

  v8i acc_hh[2] = {};
  v8i acc_hl[2] = {};
  v8i acc_lh[2] = {};
  v8i acc_ll[2] = {};

  for (int kb = 0; kb < K64; ++kb) {
    for (int t = tid; t < 512; t += nthreads) {
      int sub = t >> 8;
      int li = t & 255;
      int flane = li >> 3, r = li & 7;
      int ncol = flane & 15, hsel = flane >> 4;
      int kbase = kb * 64 + ((r & 1) << 2) + ((r >> 1) << 4) + (hsel << 3);
      int pos = (ntile << 5) + (sub << 4) + ncol;
      int b = pos >> howoShift;
      int r2 = pos & HoWomask;
      int oh = r2 >> woShift;
      int ow = r2 & Wmask;
      int hiW = 0, loW = 0;
#pragma unroll
      for (int j = 0; j < 4; ++j) {
        int kk = kbase + j;
        int cin = kk / FF;
        int rem = kk - cin * FF;
        int fh = rem / F;
        int fw = rem - fh * F;
        int ih = oh * stride - pad + fh;
        int iw = ow * stride - pad + fw;
        bool inb = (kk < K) & (ih >= 0) & (ih < H) & (iw >= 0) & (iw < W);
        int idx = ((b * Cin + cin) * H + ih) * W + iw;
        int q = (int)qin[inb ? idx : 0];
        q = inb ? q : 0;
        hiW |= ((q >> 8) & 0xFF) << (8 * j);
        loW |= (q & 0xFF) << (8 * j);
      }
      sHi[t] = hiW;
      sLo[t] = loW;
    }
    __syncthreads();

    const int base = ((mtile * K64 + kb) * 32 + lane) * 8;
    v8i a_h = *(const v8i*)(a_hi + base);
    v8i a_l = *(const v8i*)(a_lo + base);

#pragma unroll
    for (int sub = 0; sub < 2; ++sub) {
      v8i b_h = *(const v8i*)(&sHi[sub * 256 + lane * 8]);
      v8i b_l = *(const v8i*)(&sLo[sub * 256 + lane * 8]);
      acc_hh[sub] = __builtin_amdgcn_wmma_i32_16x16x64_iu8(true,  a_h, true,  b_h, acc_hh[sub], false, false);
      acc_hl[sub] = __builtin_amdgcn_wmma_i32_16x16x64_iu8(true,  a_h, false, b_l, acc_hl[sub], false, false);
      acc_lh[sub] = __builtin_amdgcn_wmma_i32_16x16x64_iu8(false, a_l, true,  b_h, acc_lh[sub], false, false);
      acc_ll[sub] = __builtin_amdgcn_wmma_i32_16x16x64_iu8(false, a_l, false, b_l, acc_ll[sub], false, false);
    }
    __syncthreads();
  }

  const int ncol = lane & 15;
  const int hsel = lane >> 4;
#pragma unroll
  for (int sub = 0; sub < 2; ++sub) {
    int pos = (ntile << 5) + (sub << 4) + ncol;
    int b = pos >> howoShift;
    int r2 = pos & HoWomask;
#pragma unroll
    for (int v = 0; v < 8; ++v) {
      int c = mtile * 16 + v + 8 * hsel;
      float y = ((float)acc_hh[sub][v]) * 65536.0f
              + ((float)(acc_hl[sub][v] + acc_lh[sub][v])) * 256.0f
              + (float)acc_ll[sub][v];
      y *= (1.0f / 67108864.0f);
      float qf = fminf(fmaxf(rintf(y * 16777216.0f), -2147483648.0f), 2147483647.0f);
      y = qf * (1.0f / 16777216.0f);
      y = y * bn_inv[c] + bn_sh[c];
      size_t o = ((size_t)(b * Cout + c) << howoShift) + r2;
      if (resid) y += resid[o];
      if (do_relu) y = fmaxf(y, 0.0f);
      outp[o] = y;
      qout[o] = quant16_13(y);
    }
  }
}

__global__ void pool_kernel(const float* __restrict__ in, float* __restrict__ pooled,
                            int C, int HW) {
  int i = blockIdx.x * blockDim.x + threadIdx.x;
  if (i >= BATCH * C) return;
  const float* p = in + (size_t)i * HW;
  float s = 0.0f;
  for (int j = 0; j < HW; ++j) s += p[j];
  pooled[i] = s / (float)HW;
}

__global__ void fc_kernel(const float* __restrict__ pooled, const float* __restrict__ w,
                          const float* __restrict__ bias, float* __restrict__ out) {
  int i = blockIdx.x * blockDim.x + threadIdx.x;
  if (i >= BATCH * 10) return;
  int b = i / 10, j = i - b * 10;
  float s = bias[j];
  const float* p = pooled + b * 64;
  const float* ww = w + j * 64;
  for (int c = 0; c < 64; ++c) s += p[c] * ww[c];
  out[i] = s;
}

struct Cfg { int widx, Cin, Cout, H, W, stride, pad, F; };

static inline int ilog2(int x) { return 31 - __builtin_clz(x); }

extern "C" void kernel_launch(void* const* d_in, const int* in_sizes, int n_in,
                              void* d_out, int out_size, void* d_ws, size_t ws_size,
                              hipStream_t stream) {
  (void)in_sizes; (void)n_in; (void)out_size; (void)ws_size;

  static const Cfg cfg[22] = {
    {  1,  3, 16, 32, 32, 1, 1, 3},
    {  6, 16, 16, 32, 32, 1, 0, 1},
    { 11, 16, 16, 32, 32, 1, 1, 3},
    { 16, 16, 16, 32, 32, 1, 1, 3},
    { 21, 16, 16, 32, 32, 1, 1, 3},
    { 26, 16, 16, 32, 32, 1, 1, 3},
    { 31, 16, 16, 32, 32, 1, 1, 3},
    { 36, 16, 16, 32, 32, 1, 1, 3},
    { 41, 16, 32, 32, 32, 2, 0, 1},
    { 46, 16, 32, 32, 32, 2, 1, 3},
    { 51, 32, 32, 16, 16, 1, 1, 3},
    { 56, 32, 32, 16, 16, 1, 1, 3},
    { 61, 32, 32, 16, 16, 1, 1, 3},
    { 66, 32, 32, 16, 16, 1, 1, 3},
    { 71, 32, 32, 16, 16, 1, 1, 3},
    { 76, 32, 64, 16, 16, 2, 0, 1},
    { 81, 32, 64, 16, 16, 2, 1, 3},
    { 86, 64, 64,  8,  8, 1, 1, 3},
    { 91, 64, 64,  8,  8, 1, 1, 3},
    { 96, 64, 64,  8,  8, 1, 1, 3},
    {101, 64, 64,  8,  8, 1, 1, 3},
    {106, 64, 64,  8,  8, 1, 1, 3},
  };

  char* wsb = (char*)d_ws;
  const size_t MB = 1024 * 1024;
  float* bufX = (float*)(wsb + 0);
  float* buf[3] = { (float*)(wsb + 12 * MB),
                    (float*)(wsb + 76 * MB),
                    (float*)(wsb + 140 * MB) };
  float* bufR = (float*)(wsb + 204 * MB);
  float* pooled = (float*)(wsb + 268 * MB);
  short* qX = (short*)(wsb + 269 * MB);
  short* qbuf[3] = { (short*)(wsb + 276 * MB),
                     (short*)(wsb + 308 * MB),
                     (short*)(wsb + 340 * MB) };
  short* qR = (short*)(wsb + 372 * MB);
  char* wpack = wsb + 405 * MB;
  const size_t LAYER_BYTES = 80 * 1024;

  auto run_conv = [&](const short* qinp, int li, const float* resid, int relu,
                      float* outp, short* qoutp) {
    const Cfg& c = cfg[li];
    int K = c.Cin * c.F * c.F;
    int K64 = (K + 63) / 64;
    int mtiles = c.Cout >> 4;
    int Ho = (c.H + 2 * c.pad - c.F) / c.stride + 1;
    int Wo = (c.W + 2 * c.pad - c.F) / c.stride + 1;
    char* base = wpack + (size_t)li * LAYER_BYTES;
    int* a_hi = (int*)base;
    int* a_lo = (int*)(base + 37120);
    float* bn_inv = (float*)(base + 74240);
    float* bn_sh  = (float*)(base + 74752);

    int ptotal = mtiles * K64 * 256;
    pack_weights_kernel<<<dim3((ptotal + 255) / 256), dim3(256), 0, stream>>>(
        (const float*)d_in[c.widx], c.Cout, K, K64, a_hi, a_lo);
    bn_prep_kernel<<<dim3(1), dim3(64), 0, stream>>>(
        (const float*)d_in[c.widx + 1], (const float*)d_in[c.widx + 2],
        (const float*)d_in[c.widx + 3], (const float*)d_in[c.widx + 4],
        bn_inv, bn_sh, c.Cout);

    int Npos = BATCH * Ho * Wo;
    int woShift = ilog2(Wo);
    int howoShift = ilog2(Ho * Wo);
    dim3 grid(Npos / 32);
    dim3 block(32 * mtiles);
    if (c.F == 3) {
      conv_bn_wmma_kernel<3><<<grid, block, 0, stream>>>(
          qinp, a_hi, a_lo, bn_inv, bn_sh, resid, outp, qoutp,
          c.Cin, c.Cout, c.H, c.W, woShift, howoShift, c.stride, c.pad, K, K64, relu);
    } else {
      conv_bn_wmma_kernel<1><<<grid, block, 0, stream>>>(
          qinp, a_hi, a_lo, bn_inv, bn_sh, resid, outp, qoutp,
          c.Cin, c.Cout, c.H, c.W, woShift, howoShift, c.stride, c.pad, K, K64, relu);
    }
  };

  {
    int total = BATCH * 3 * 32 * 32;
    normalize_kernel<<<dim3((total + 255) / 256), dim3(256), 0, stream>>>(
        (const float*)d_in[0], bufX, qX, total);
  }

  int cur = 0;
  run_conv(qX, 0, nullptr, 1, buf[0], qbuf[0]);

  int li = 1;
  for (int s = 0; s < 3; ++s) {
    run_conv(qbuf[cur], li + 0, nullptr, 0, bufR, qR);
    int t1 = (cur + 1) % 3, t2 = (cur + 2) % 3;
    run_conv(qbuf[cur], li + 1, nullptr, 1, buf[t1], qbuf[t1]);
    run_conv(qbuf[t1], li + 2, bufR, 1, buf[t2], qbuf[t2]);
    cur = t2;
    for (int bblk = 0; bblk < 2; ++bblk) {
      int u1 = (cur + 1) % 3, u2 = (cur + 2) % 3;
      run_conv(qbuf[cur], li + 3 + bblk * 2, nullptr, 1, buf[u1], qbuf[u1]);
      run_conv(qbuf[u1], li + 4 + bblk * 2, buf[cur], 1, buf[u2], qbuf[u2]);
      cur = u2;
    }
    li += 7;
  }

  pool_kernel<<<dim3((BATCH * 64 + 255) / 256), dim3(256), 0, stream>>>(
      buf[cur], pooled, 64, 64);
  fc_kernel<<<dim3((BATCH * 10 + 255) / 256), dim3(256), 0, stream>>>(
      pooled, (const float*)d_in[111], (const float*)d_in[112], (float*)d_out);
}
